// PluckerAttention_60739427500728
// MI455X (gfx1250) — matrix operand order repaired and checked by reading, not compiled, not run on hardware
//
#include <hip/hip_runtime.h>
#include <hip/hip_bf16.h>
#include <math.h>

// ---------------------------------------------------------------------------
// PluckerAttention for MI455X (gfx1250, wave32, WMMA + double-buffered TDM)
// B=2, T=2048, D=1024, H=16, d_head=64
// ---------------------------------------------------------------------------

typedef __attribute__((ext_vector_type(16))) _Float16 v16h;
typedef __attribute__((ext_vector_type(8)))  _Float16 v8h;
typedef __attribute__((ext_vector_type(8)))  float    v8f;

#define B_  2
#define T_  2048
#define D_  1024
#define H_  16
#define DH_ 64
#define M_  (B_ * T_)   // 4096 rows
#define P_  64          // 4*H columns of W1/W2
#define LDH 40          // padded LDS row pitch in halves (80B: 16B-aligned, bank-spread)
#define TILE_BYTES (64 * LDH * 2)   // one 64-row staged tile

// ---- Tensor Data Mover (guarded: compiles on both toolchains) -------------

#if defined(__AMDGCN__) && __has_builtin(__builtin_amdgcn_tensor_load_to_lds) && \
    __has_builtin(__builtin_amdgcn_s_wait_tensorcnt)
#define HAVE_TDM 1
#define NBUF 2
#else
#define HAVE_TDM 0
#define NBUF 1
#endif

#if HAVE_TDM
typedef unsigned int u32x4 __attribute__((ext_vector_type(4)));
typedef int          i32x4 __attribute__((ext_vector_type(4)));
typedef int          i32x8 __attribute__((ext_vector_type(8)));

// 2D f16 tile load: 32 halves (64B) contiguous per row, tile_d1 rows, row
// stride stride0 elems; LDS gets +16B pad per 64B row (pad_interval=3 -> 64B,
// pad_amount=3 -> 4 DWORDs) to match the LDH=40 pitch.
__device__ __forceinline__ void tdm_load_2d_f16(unsigned lds_off, const void* gp,
                                                unsigned tile_d0, unsigned tile_d1,
                                                unsigned tensor_d0, unsigned tensor_d1,
                                                unsigned stride0) {
  unsigned long long ga = (unsigned long long)(size_t)gp;
  u32x4 g0;
  g0.x = 1u;                                    // count=1, user descriptor
  g0.y = lds_off;                               // lds_addr
  g0.z = (unsigned)(ga & 0xFFFFFFFFu);          // global_addr[31:0]
  g0.w = (unsigned)((ga >> 32) & 0x01FFFFFFu) | (2u << 30);   // addr[56:32] | type=2
  i32x8 g1;
  g1[0] = (int)((1u << 16) | (1u << 20) | (3u << 22) | (3u << 25)); // 2B elems, pad on
  g1[1] = (int)((tensor_d0 & 0xFFFFu) << 16);                       // tensor_dim0[15:0]
  g1[2] = (int)(((tensor_d0 >> 16) & 0xFFFFu) | ((tensor_d1 & 0xFFFFu) << 16));
  g1[3] = (int)(((tensor_d1 >> 16) & 0xFFFFu) | ((tile_d0 & 0xFFFFu) << 16));
  g1[4] = (int)(tile_d1 & 0xFFFFu);                                 // tile_dim1, dim2=0
  g1[5] = (int)stride0;                                             // dim0_stride[31:0]
  g1[6] = 0;
  g1[7] = 0;
  i32x4 z4 = {0, 0, 0, 0};
#if __clang_major__ >= 23
  i32x8 z8 = {0, 0, 0, 0, 0, 0, 0, 0};
  __builtin_amdgcn_tensor_load_to_lds(g0, g1, z4, z4, z8, 0);
#else
  __builtin_amdgcn_tensor_load_to_lds(g0, g1, z4, z4, 0);
#endif
}
#endif  // HAVE_TDM

// ---- WMMA fragment loaders (ISA 7.12.2 layouts, wave32) -------------------

// A 16x32 f16: lane L -> row m=L&15; halves {g*8+i} and {16+g*8+i}, g=L>>4.
__device__ __forceinline__ v16h load_a_frag(const _Float16* p, int lane) {
  const int m = lane & 15, g = lane >> 4;
  const v8h lo = *(const v8h*)&p[m * LDH + g * 8];
  const v8h hi = *(const v8h*)&p[m * LDH + 16 + g * 8];
  return __builtin_shufflevector(lo, hi, 0, 1, 2, 3, 4, 5, 6, 7,
                                 8, 9, 10, 11, 12, 13, 14, 15);
}

// B 32x16 f16 from N-major LDS tile Bs[n][k]: lane L -> col n=L&15,
// halves {g*8+i} and {16+g*8+i}, g=L>>4.
__device__ __forceinline__ v16h load_bT_frag(const _Float16* p, int lane) {
  const int n = lane & 15, g = lane >> 4;
  const v8h lo = *(const v8h*)&p[n * LDH + g * 8];
  const v8h hi = *(const v8h*)&p[n * LDH + 16 + g * 8];
  return __builtin_shufflevector(lo, hi, 0, 1, 2, 3, 4, 5, 6, 7,
                                 8, 9, 10, 11, 12, 13, 14, 15);
}

// ---- conversions ----------------------------------------------------------

__global__ void f32_to_f16_k(const float* __restrict__ in,
                             _Float16* __restrict__ out, int n) {
  int i = blockIdx.x * blockDim.x + threadIdx.x;
  if (i < n) out[i] = (_Float16)in[i];
}

// W (K x N, f32) -> WT (N x K, f16), 32x32 LDS tiles, 256 threads.
__global__ void f32_to_f16_tr_k(const float* __restrict__ in,
                                _Float16* __restrict__ out, int K, int N) {
  __shared__ _Float16 tile[32][33];
  const int k0 = blockIdx.x * 32, n0 = blockIdx.y * 32;
  const int tx = threadIdx.x & 31, ty = threadIdx.x >> 5;   // 32 x 8
#pragma unroll
  for (int i = ty; i < 32; i += 8)
    tile[i][tx] = (_Float16)in[(size_t)(k0 + i) * N + n0 + tx];
  __syncthreads();
#pragma unroll
  for (int i = ty; i < 32; i += 8)
    out[(size_t)(n0 + i) * K + k0 + tx] = tile[tx][i];
}

// ---- WMMA GEMM: C[M,N] = A[M,K] @ BT[N,K]^T (+bias) -----------------------
// Block 256 threads (8 waves, 2x4). Block tile 64x64; 2 v_wmma per wave per
// K-step. Both operands staged by TDM (double-buffered, wave 0 issues, one
// barrier per K-step); manual b128 fallback when TDM builtin is absent.
// Optional CvT epilogue scatters V into head-major transposed (B*H*DH, T).

__global__ __launch_bounds__(256)
void gemm_f16_wmma(const _Float16* __restrict__ A,     // M x K
                   const _Float16* __restrict__ BT,    // N x K (pre-transposed)
                   const float* __restrict__ bias,
                   float* __restrict__ Cf32, _Float16* __restrict__ Cf16,
                   _Float16* __restrict__ CvT,
                   int M, int N, int K) {
  __shared__ __align__(16) _Float16 As[NBUF][64][LDH];
  __shared__ __align__(16) _Float16 Bs[NBUF][64][LDH];

  const int tid  = threadIdx.x;
  const int lane = tid & 31;
  const int wave = tid >> 5;
  const int wr   = wave >> 2;   // 0..1
  const int wc   = wave & 3;    // 0..3
  const int row0 = blockIdx.x * 64;
  const int col0 = blockIdx.y * 64;

  v8f acc0 = {};
  v8f acc1 = {};

#if HAVE_TDM
  const unsigned ldsA0 = (unsigned)(size_t)&As[0][0][0];
  const unsigned ldsB0 = (unsigned)(size_t)&Bs[0][0][0];

  if (wave == 0) {
    tdm_load_2d_f16(ldsA0, A + (size_t)row0 * K, 32, 64,
                    (unsigned)K, (unsigned)M, (unsigned)K);
    tdm_load_2d_f16(ldsB0, BT + (size_t)col0 * K, 32, 64,
                    (unsigned)K, (unsigned)N, (unsigned)K);
    __builtin_amdgcn_s_wait_tensorcnt(0);
  }
  __syncthreads();

  int cur = 0;
  for (int kk = 0; kk < K; kk += 32, cur ^= 1) {
    if (wave == 0 && kk + 32 < K) {     // prefetch next K-step into other buffer
      unsigned nb = (unsigned)(cur ^ 1) * (unsigned)TILE_BYTES;
      tdm_load_2d_f16(ldsA0 + nb, A + (size_t)row0 * K + kk + 32, 32, 64,
                      (unsigned)K, (unsigned)M, (unsigned)K);
      tdm_load_2d_f16(ldsB0 + nb, BT + (size_t)col0 * K + kk + 32, 32, 64,
                      (unsigned)K, (unsigned)N, (unsigned)K);
    }
    v16h bf = load_bT_frag(&Bs[cur][wc * 16][0], lane);
    v16h a0 = load_a_frag(&As[cur][wr * 16][0], lane);
    v16h a1 = load_a_frag(&As[cur][32 + wr * 16][0], lane);
    acc0 = __builtin_amdgcn_wmma_f32_16x16x32_f16(false, a0, false, bf,
                                                  (short)0, acc0, false, false);
    acc1 = __builtin_amdgcn_wmma_f32_16x16x32_f16(false, a1, false, bf,
                                                  (short)0, acc1, false, false);
    if (wave == 0) __builtin_amdgcn_s_wait_tensorcnt(0);
    __syncthreads();
  }
#else
  const int sr = tid >> 2;              // 64 rows, 4 b128 chunks per row
  const int sc = (tid & 3) * 8;
  for (int kk = 0; kk < K; kk += 32) {
    *(v8h*)&As[0][sr][sc] = *(const v8h*)&A[(size_t)(row0 + sr) * K + kk + sc];
    *(v8h*)&Bs[0][sr][sc] = *(const v8h*)&BT[(size_t)(col0 + sr) * K + kk + sc];
    if (kk + 32 < K)
      __builtin_prefetch(&A[(size_t)(row0 + sr) * K + kk + 32 + sc], 0, 3);
    __syncthreads();
    v16h bf = load_bT_frag(&Bs[0][wc * 16][0], lane);
    v16h a0 = load_a_frag(&As[0][wr * 16][0], lane);
    v16h a1 = load_a_frag(&As[0][32 + wr * 16][0], lane);
    acc0 = __builtin_amdgcn_wmma_f32_16x16x32_f16(false, a0, false, bf,
                                                  (short)0, acc0, false, false);
    acc1 = __builtin_amdgcn_wmma_f32_16x16x32_f16(false, a1, false, bf,
                                                  (short)0, acc1, false, false);
    __syncthreads();
  }
#endif

  const int n    = lane & 15;
  const int colg = col0 + wc * 16 + n;
  const float bb = bias ? bias[colg] : 0.0f;
#pragma unroll
  for (int r = 0; r < 8; ++r) {
    int m0 = row0 + wr * 16 + (lane >> 4) * 8 + r;
    int m1 = m0 + 32;
    float va = acc0[r] + bb;
    float vb = acc1[r] + bb;
    if (Cf32) { Cf32[(size_t)m0 * N + colg] = va;
                Cf32[(size_t)m1 * N + colg] = vb; }
    if (Cf16) { Cf16[(size_t)m0 * N + colg] = (_Float16)va;
                Cf16[(size_t)m1 * N + colg] = (_Float16)vb; }
    if (CvT) {  // scatter to head-major transposed V: (B*H*DH, T)
      int hh = colg >> 6, dd = colg & (DH_ - 1);
      int b0 = m0 >> 11, t0 = m0 & (T_ - 1);
      int b1 = m1 >> 11, t1 = m1 & (T_ - 1);
      CvT[((size_t)((b0 * H_ + hh) * DH_ + dd)) * T_ + t0] = (_Float16)va;
      CvT[((size_t)((b1 * H_ + hh) * DH_ + dd)) * T_ + t1] = (_Float16)vb;
    }
  }
}

// ---- Plücker lines: p1,p2 (M x 64 f32) -> normalized 6-vectors ------------
// Output head-major: lines[((b*H + h)*T + t)*6 + j]

__global__ void plucker_lines_k(const float* __restrict__ p1,
                                const float* __restrict__ p2,
                                float* __restrict__ lines) {
  int idx = blockIdx.x * blockDim.x + threadIdx.x;   // over M_*H_
  if (idx >= M_ * H_) return;
  int h = idx & (H_ - 1);
  int row = idx >> 4;
  const float* a = p1 + (size_t)row * P_ + h * 4;
  const float* c = p2 + (size_t)row * P_ + h * 4;
  float l0 = a[0] * c[1] - a[1] * c[0];
  float l1 = a[0] * c[2] - a[2] * c[0];
  float l2 = a[0] * c[3] - a[3] * c[0];
  float l3 = a[1] * c[2] - a[2] * c[1];
  float l4 = a[1] * c[3] - a[3] * c[1];
  float l5 = a[2] * c[3] - a[3] * c[2];
  float nrm = sqrtf(l0*l0 + l1*l1 + l2*l2 + l3*l3 + l4*l4 + l5*l5);
  float inv = 1.0f / fmaxf(nrm, 1e-12f);
  int b = row >> 11;
  int t = row & (T_ - 1);
  float* o = lines + ((size_t)(b * H_ + h) * T_ + t) * 6;
  o[0] = l0 * inv; o[1] = l1 * inv; o[2] = l2 * inv;
  o[3] = l3 * inv; o[4] = l4 * inv; o[5] = l5 * inv;
}

// ---- Causal Plücker attention (flash-style, WMMA for attn@v) --------------
// grid (T/32, B*H); block 256. V is head-major transposed (B*H*DH, T) so the
// V tile is a clean 2D TDM tile; its DMA overlaps the fp32 score + softmax.

__global__ __launch_bounds__(256)
void plucker_attn_k(const float* __restrict__ lines,
                    const _Float16* __restrict__ vT,      // (B*H*DH, T) f16
                    const float* __restrict__ log_temp,
                    _Float16* __restrict__ outp) {        // (M_, D_) f16, pre-Wo
  __shared__ float    Lq[32][6];
  __shared__ float    Lk[32][6];
  __shared__ float    S[32][32];
  __shared__ __align__(16) _Float16 Pa[32][LDH];
  __shared__ __align__(16) _Float16 VtT[64][LDH];
  __shared__ float    rowM[32];
  __shared__ float    rowL[32];
  __shared__ float    rowScale[32];

  const int tid  = threadIdx.x;
  const int lane = tid & 31;
  const int wave = tid >> 5;
  const int wr   = wave >> 2;
  const int wc   = wave & 3;

  const int bh  = blockIdx.y;
  const int b   = bh >> 4;
  const int h   = bh & (H_ - 1);
  const int qt0 = blockIdx.x * 32;
  const float temp = expf(log_temp[h]);
  const float NEG_INF = -__builtin_inff();

  for (int i = tid; i < 32 * 6; i += 256) {
    int r = i / 6, c = i % 6;
    Lq[r][c] = lines[((size_t)bh * T_ + qt0 + r) * 6 + c];
  }
  if (tid < 32) { rowM[tid] = NEG_INF; rowL[tid] = 0.0f; }

#if HAVE_TDM
  const unsigned ldsV = (unsigned)(size_t)&VtT[0][0];
#endif

  v8f acc = {};
  const int nkt = (qt0 >> 5) + 1;

  for (int kt = 0; kt < nkt; ++kt) {
    const int u0 = kt * 32;
    // V tile (64 dims x 32 keys): DMA overlaps score + softmax phases
#if HAVE_TDM
    if (wave == 0)
      tdm_load_2d_f16(ldsV, vT + (size_t)(bh * DH_) * T_ + u0, 32, 64,
                      (unsigned)T_, (unsigned)(B_ * H_ * DH_), (unsigned)T_);
#else
    {
      int d = tid >> 2, c = (tid & 3) * 8;
      *(v8h*)&VtT[d][c] = *(const v8h*)&vT[(size_t)(bh * DH_ + d) * T_ + u0 + c];
    }
#endif
    for (int i = tid; i < 32 * 6; i += 256) {
      int r = i / 6, c = i % 6;
      Lk[r][c] = lines[((size_t)bh * T_ + u0 + r) * 6 + c];
    }
    if (kt + 1 < nkt)
      __builtin_prefetch(&lines[((size_t)bh * T_ + u0 + 32) * 6], 0, 2);
    __syncthreads();

    // fp32 incidence scores + causal mask + log transform
#pragma unroll
    for (int i = tid; i < 32 * 32; i += 256) {
      int q = i >> 5, u = i & 31;
      int tq = qt0 + q, tu = u0 + u;
      float s;
      if (tu > tq) {
        s = NEG_INF;
      } else {
        const float* a = Lq[q];
        const float* c = Lk[u];
        float inc = a[0]*c[5] - a[1]*c[4] + a[2]*c[3]
                  + a[3]*c[2] - a[4]*c[1] + a[5]*c[0];
        s = -temp * logf(fabsf(inc) + 1e-8f);
      }
      S[q][u] = s;
    }
    __syncthreads();

    // online softmax per query row (wave 0, one lane per row)
    if (tid < 32) {
      const int q = tid;
      float mx = rowM[q];
#pragma unroll
      for (int u = 0; u < 32; ++u) mx = fmaxf(mx, S[q][u]);
      float scale = expf(rowM[q] - mx);
      float sum = 0.0f;
#pragma unroll
      for (int u = 0; u < 32; ++u) {
        float p = expf(S[q][u] - mx);
        Pa[q][u] = (_Float16)p;
        sum += p;
      }
      rowL[q]     = rowL[q] * scale + sum;
      rowScale[q] = scale;
      rowM[q]     = mx;
    }
#if HAVE_TDM
    if (wave == 0) __builtin_amdgcn_s_wait_tensorcnt(0);
#endif
    __syncthreads();

    // rescale accumulator, then acc += P @ V via WMMA
#pragma unroll
    for (int r = 0; r < 8; ++r) {
      int row = wr * 16 + (lane >> 4) * 8 + r;
      acc[r] *= rowScale[row];
    }
    v16h af = load_a_frag(&Pa[wr * 16][0], lane);
    v16h bf = load_bT_frag(&VtT[wc * 16][0], lane);
    acc = __builtin_amdgcn_wmma_f32_16x16x32_f16(false, af, false, bf,
                                                 (short)0, acc, false, false);
    __syncthreads();
  }

  const int col = wc * 16 + (lane & 15);
#pragma unroll
  for (int r = 0; r < 8; ++r) {
    int row = wr * 16 + (lane >> 4) * 8 + r;
    float val = acc[r] / rowL[row];
    outp[(size_t)(b * T_ + qt0 + row) * D_ + h * DH_ + col] = (_Float16)val;
  }
}

// ---------------------------------------------------------------------------

extern "C" void kernel_launch(void* const* d_in, const int* in_sizes, int n_in,
                              void* d_out, int out_size, void* d_ws, size_t ws_size,
                              hipStream_t stream) {
  (void)in_sizes; (void)n_in; (void)out_size; (void)ws_size;

  const float* x        = (const float*)d_in[0];   // (B,T,D)
  const float* W1       = (const float*)d_in[1];   // (D,64)
  const float* W2       = (const float*)d_in[2];   // (D,64)
  const float* log_temp = (const float*)d_in[3];   // (H,)
  const float* Wv       = (const float*)d_in[4];   // (D,D)
  const float* bv       = (const float*)d_in[5];   // (D,)
  const float* Wo       = (const float*)d_in[6];   // (D,D)
  const float* bo       = (const float*)d_in[7];   // (D,)

  char* ws = (char*)d_ws;
  size_t off = 0;
  auto carve = [&](size_t bytes) -> void* {
    void* p = ws + off;
    off = (off + bytes + 255) & ~(size_t)255;
    return p;
  };

  _Float16* xh    = (_Float16*)carve((size_t)M_ * D_ * 2);
  _Float16* W1t   = (_Float16*)carve((size_t)D_ * P_ * 2);   // (64, 1024)
  _Float16* W2t   = (_Float16*)carve((size_t)D_ * P_ * 2);
  _Float16* Wvt   = (_Float16*)carve((size_t)D_ * D_ * 2);   // (1024, 1024)
  _Float16* Wot   = (_Float16*)carve((size_t)D_ * D_ * 2);
  float*    p1    = (float*)   carve((size_t)M_ * P_ * 4);
  float*    p2    = (float*)   carve((size_t)M_ * P_ * 4);
  float*    lines = (float*)   carve((size_t)B_ * H_ * T_ * 6 * 4);
  _Float16* vT    = (_Float16*)carve((size_t)B_ * H_ * DH_ * T_ * 2);
  _Float16* outp  = (_Float16*)carve((size_t)M_ * D_ * 2);

  // staging: x -> f16 row-major; weights -> f16 transposed (N, K)
  f32_to_f16_k<<<(M_ * D_ + 255) / 256, 256, 0, stream>>>(x, xh, M_ * D_);
  f32_to_f16_tr_k<<<dim3(D_ / 32, P_ / 32), 256, 0, stream>>>(W1, W1t, D_, P_);
  f32_to_f16_tr_k<<<dim3(D_ / 32, P_ / 32), 256, 0, stream>>>(W2, W2t, D_, P_);
  f32_to_f16_tr_k<<<dim3(D_ / 32, D_ / 32), 256, 0, stream>>>(Wv, Wvt, D_, D_);
  f32_to_f16_tr_k<<<dim3(D_ / 32, D_ / 32), 256, 0, stream>>>(Wo, Wot, D_, D_);

  // p1 = x@W1, p2 = x@W2 (f32 out)
  gemm_f16_wmma<<<dim3(M_ / 64, P_ / 64), 256, 0, stream>>>(
      xh, W1t, nullptr, p1, nullptr, nullptr, M_, P_, D_);
  gemm_f16_wmma<<<dim3(M_ / 64, P_ / 64), 256, 0, stream>>>(
      xh, W2t, nullptr, p2, nullptr, nullptr, M_, P_, D_);

  // normalized Plücker lines (head-major)
  plucker_lines_k<<<(M_ * H_ + 255) / 256, 256, 0, stream>>>(p1, p2, lines);

  // v = x@Wv + bv, scattered directly into head-major transposed (B*H*DH, T)
  gemm_f16_wmma<<<dim3(M_ / 64, D_ / 64), 256, 0, stream>>>(
      xh, Wvt, bv, nullptr, nullptr, vT, M_, D_, D_);

  // causal incidence attention, per (b,h)
  plucker_attn_k<<<dim3(T_ / 32, B_ * H_), 256, 0, stream>>>(
      lines, vT, log_temp, outp);

  // final projection: out = outp@Wo + bo (f32 out)
  gemm_f16_wmma<<<dim3(M_ / 64, D_ / 64), 256, 0, stream>>>(
      outp, Wot, bo, (float*)d_out, nullptr, nullptr, M_, D_, D_);
}
